// ChannelWiseFloat8GroupedLinear_22428319220571
// MI455X (gfx1250) — 8-bit matrix operand order repaired and checked by reading, compiled, not run on hardware
//
#include <hip/hip_runtime.h>
#include <hip/hip_bf16.h>

typedef __attribute__((ext_vector_type(16))) int   v16i;
typedef __attribute__((ext_vector_type(8)))  float v8f;
typedef int v4i_ __attribute__((vector_size(16)));

#define FP8_MAX 448.0f
#define DIN 1024
#define LDS_STRIDE 144

#define GLOBAL_AS __attribute__((address_space(1)))
#define LDS_AS    __attribute__((address_space(3)))


__device__ __forceinline__ unsigned char enc_e4m3(float v) {
  unsigned int s = (__float_as_uint(v) >> 24) & 0x80u;
  float a = fminf(fabsf(v), FP8_MAX);
  if (!(a >= 0.0009765625f))
    return (unsigned char)s;
  if (a < 0.015625f) {
    int m = (int)rintf(a * 512.0f);
    return (unsigned char)(s | (unsigned)m);
  }
  unsigned int au = __float_as_uint(a);
  unsigned int lsb = (au >> 20) & 1u;
  au += 0x0007FFFFu + lsb;
  int e = (int)((au >> 23) & 0xFFu) - 127;
  unsigned int mant = (au >> 20) & 7u;
  if (e > 8) { e = 8; mant = 6; }
  return (unsigned char)(s | ((unsigned)(e + 7) << 3) | mant);
}

__device__ __forceinline__ unsigned int pack4_e4m3(float a, float b, float c, float d) {
  a = fminf(fmaxf(a, -FP8_MAX), FP8_MAX);
  b = fminf(fmaxf(b, -FP8_MAX), FP8_MAX);
  c = fminf(fmaxf(c, -FP8_MAX), FP8_MAX);
  d = fminf(fmaxf(d, -FP8_MAX), FP8_MAX);
#if __has_builtin(__builtin_amdgcn_cvt_pk_fp8_f32)
  int p = 0;
  p = __builtin_amdgcn_cvt_pk_fp8_f32(a, b, p, false);
  p = __builtin_amdgcn_cvt_pk_fp8_f32(c, d, p, true);
  return (unsigned int)p;
#else
  return (unsigned int)enc_e4m3(a)         | ((unsigned int)enc_e4m3(b) << 8) |
         ((unsigned int)enc_e4m3(c) << 16) | ((unsigned int)enc_e4m3(d) << 24);
#endif
}


__device__ __forceinline__ void async_copy16(const unsigned char* g, unsigned char* l) {
#if __has_builtin(__builtin_amdgcn_global_load_async_to_lds_b128)
  __builtin_amdgcn_global_load_async_to_lds_b128(
      (GLOBAL_AS v4i_*)g, (LDS_AS v4i_*)l, 0, 0);
#else
  *(int4*)l = *(const int4*)g;
#endif
}

__device__ __forceinline__ void wait_async0() {
#if __has_builtin(__builtin_amdgcn_s_wait_asynccnt)
  __builtin_amdgcn_s_wait_asynccnt(0);
#else
  asm volatile("s_wait_asynccnt 0x0" ::: "memory");
#endif
}

__device__ __forceinline__ float block_amax256(float a, float* smem8) {
  #pragma unroll
  for (int off = 16; off; off >>= 1) a = fmaxf(a, __shfl_xor(a, off, 32));
  if ((threadIdx.x & 31) == 0) smem8[threadIdx.x >> 5] = a;
  __syncthreads();
  float m = smem8[0];
  #pragma unroll
  for (int i = 1; i < 8; ++i) m = fmaxf(m, smem8[i]);
  return m;
}

__global__ __launch_bounds__(256)
void xquant_kernel(const float* __restrict__ x, unsigned char* __restrict__ xq,
                   float* __restrict__ sx) {
  __shared__ float red[8];
  const int row = blockIdx.x;
  const int tid = threadIdx.x;
  const float* xr = x + (size_t)row * DIN;
  float4 v = *(const float4*)(xr + tid * 4);
  float a = fmaxf(fmaxf(fabsf(v.x), fabsf(v.y)), fmaxf(fabsf(v.z), fabsf(v.w)));
  float amax = block_amax256(a, red);
  float scale = fmaxf(amax, 1e-12f) * (1.0f / FP8_MAX);
  float r = 1.0f / scale;
  unsigned int packed = pack4_e4m3(v.x * r, v.y * r, v.z * r, v.w * r);
  *(unsigned int*)(xq + (size_t)row * DIN + tid * 4) = packed;
  if (tid == 0) sx[row] = scale;
}

__global__ __launch_bounds__(256)
void wamax_kernel(const float* __restrict__ w, float* __restrict__ amax,
                  int per_expert) {
  __shared__ float red[8];
  const int e = blockIdx.x >> 6;
  const int chunk = blockIdx.x & 63;
  const int n = per_expert >> 6;
  const float* base = w + (size_t)e * per_expert + (size_t)chunk * n;
  float a = 0.0f;
  for (int i = threadIdx.x * 4; i < n; i += 256 * 4) {
    float4 v = *(const float4*)(base + i);
    a = fmaxf(a, fmaxf(fmaxf(fabsf(v.x), fabsf(v.y)), fmaxf(fabsf(v.z), fabsf(v.w))));
  }
  float m = block_amax256(a, red);
  if (threadIdx.x == 0)
    atomicMax((unsigned int*)(amax + e), __float_as_uint(m));
}

__global__ __launch_bounds__(256)
void wquant_kernel(const float* __restrict__ w, unsigned char* __restrict__ wq,
                   const float* __restrict__ amax, int per_expert) {
  size_t idx = 4ull * ((size_t)blockIdx.x * 256 + threadIdx.x);
  int e = (int)(idx / (size_t)per_expert);
  float r = FP8_MAX / fmaxf(amax[e], 1e-12f);
  float4 v = *(const float4*)(w + idx);
  *(unsigned int*)(wq + idx) = pack4_e4m3(v.x * r, v.y * r, v.z * r, v.w * r);
}

__global__ __launch_bounds__(256)
void gemm_fp8_kernel(const unsigned char* __restrict__ xq,
                     const unsigned char* __restrict__ wq,
                     const float* __restrict__ sx,
                     const float* __restrict__ wamax,
                     const int* __restrict__ tpe,
                     __hip_bfloat16* __restrict__ out,
                     int dout, int ne) {
  __shared__ __attribute__((aligned(16))) unsigned char Asmem[2][128 * LDS_STRIDE];
  __shared__ __attribute__((aligned(16))) unsigned char Bsmem[2][128 * LDS_STRIDE];

  const int tid    = threadIdx.x;
  const int lane   = tid & 31;
  const int wave   = tid >> 5;
  const int wave_m = wave >> 1;
  const int wave_n = wave & 1;
  const int half   = lane >> 4;
  const int l16    = lane & 15;
  const int row0   = blockIdx.x * 128;
  const int col0   = blockIdx.y * 128;

  int e = ne - 1, csum = 0;
  for (int i = 0; i < ne; ++i) { csum += tpe[i]; if (row0 < csum) { e = i; break; } }

  const unsigned char* xbase = xq + (size_t)row0 * DIN;
  const unsigned char* wbase = wq + ((size_t)e * dout + col0) * DIN;

  v8f acc[2][4];
  #pragma unroll
  for (int i = 0; i < 2; ++i)
    #pragma unroll
    for (int nt = 0; nt < 4; ++nt)
      #pragma unroll
      for (int j = 0; j < 8; ++j) acc[i][nt][j] = 0.0f;

  auto stage = [&](int k0, int buf) {
    unsigned char* A = &Asmem[buf][0];
    unsigned char* B = &Bsmem[buf][0];
    #pragma unroll
    for (int i = 0; i < 4; ++i) {
      int c  = tid + (i << 8);
      int r  = c >> 3;
      int cb = (c & 7) << 4;
      async_copy16(xbase + (size_t)r * DIN + k0 + cb, A + r * LDS_STRIDE + cb);
      async_copy16(wbase + (size_t)r * DIN + k0 + cb, B + r * LDS_STRIDE + cb);
    }
  };

  auto loadB = [&](const unsigned char* B, int nt, v16i& bf) {
    const unsigned char* brow = B + (wave_n * 64 + nt * 16 + l16) * LDS_STRIDE;
    #pragma unroll
    for (int q = 0; q < 4; ++q) {
      int4 d = *(const int4*)(brow + (q << 5) + (half << 4));
      bf[4 * q] = d.x; bf[4 * q + 1] = d.y; bf[4 * q + 2] = d.z; bf[4 * q + 3] = d.w;
    }
  };

  constexpr int STEPS = DIN / 128;

  stage(0, 0);
  wait_async0();
  __syncthreads();

  for (int k = 0; k < STEPS; ++k) {
    const int cur = k & 1;
    if (k + 1 < STEPS) stage((k + 1) * 128, 1 - cur);

    const unsigned char* A = &Asmem[cur][0];
    const unsigned char* B = &Bsmem[cur][0];

    v16i af[2];
    #pragma unroll
    for (int i = 0; i < 2; ++i) {
      const unsigned char* arow = A + (wave_m * 32 + i * 16 + l16) * LDS_STRIDE;
      #pragma unroll
      for (int p = 0; p < 8; ++p) {
        int kk = ((p >> 1) << 5) + (half << 4) + ((p & 1) << 3);
        int2 d = *(const int2*)(arow + kk);
        af[i][2 * p] = d.x; af[i][2 * p + 1] = d.y;
      }
    }

    v16i bcur;
    loadB(B, 0, bcur);
    #pragma unroll
    for (int nt = 0; nt < 4; ++nt) {
      v16i bnext = bcur;
      if (nt < 3) loadB(B, nt + 1, bnext);
      acc[0][nt] = __builtin_amdgcn_wmma_f32_16x16x128_fp8_fp8(
          af[0], bcur, (short)0, acc[0][nt], false, false);
      acc[1][nt] = __builtin_amdgcn_wmma_f32_16x16x128_fp8_fp8(
          af[1], bcur, (short)0, acc[1][nt], false, false);
      bcur = bnext;
    }

    wait_async0();
    __syncthreads();
  }

  float sw = fmaxf(wamax[e], 1e-12f) * (1.0f / FP8_MAX);
  #pragma unroll
  for (int i = 0; i < 2; ++i)
    #pragma unroll
    for (int j = 0; j < 8; ++j) {
      int row = row0 + wave_m * 32 + i * 16 + half * 8 + j;
      float s = sx[row] * sw;
      #pragma unroll
      for (int nt = 0; nt < 4; ++nt) {
        int col = col0 + wave_n * 64 + nt * 16 + l16;
        out[(size_t)row * dout + col] = __float2bfloat16(acc[i][nt][j] * s);
      }
    }
}

extern "C" void kernel_launch(void* const* d_in, const int* in_sizes, int n_in,
                              void* d_out, int out_size, void* d_ws, size_t ws_size,
                              hipStream_t stream) {
  const float* x   = (const float*)d_in[0];
  const float* w   = (const float*)d_in[1];
  const int*   tpe = (const int*)d_in[2];
  __hip_bfloat16* out = (__hip_bfloat16*)d_out;

  const int din = DIN;
  const int ne  = in_sizes[2];
  const int T   = in_sizes[0] / din;
  const int dout = in_sizes[1] / (ne * din);
  const int per_expert = dout * din;

  unsigned char* ws = (unsigned char*)d_ws;
  unsigned char* xq = ws;
  unsigned char* wq = ws + (size_t)T * din;
  float* sx  = (float*)(wq + (size_t)ne * per_expert);
  float* wam = sx + T;

  (void)hipMemsetAsync(wam, 0, ne * sizeof(float), stream);

  xquant_kernel<<<T, 256, 0, stream>>>(x, xq, sx);
  wamax_kernel<<<ne * 64, 256, 0, stream>>>(w, wam, per_expert);
  wquant_kernel<<<(int)(((size_t)ne * per_expert) / 1024), 256, 0, stream>>>(
      w, wq, wam, per_expert);

  dim3 grid(T / 128, dout / 128);
  gemm_fp8_kernel<<<grid, 256, 0, stream>>>(xq, wq, sx, wam, tpe, out, dout, ne);
}
